// Head_34179349741597
// MI455X (gfx1250) — matrix operand order repaired and checked by reading, not compiled, not run on hardware
//
#include <hip/hip_runtime.h>

typedef __bf16 bf16_t;
typedef __attribute__((ext_vector_type(2)))  __bf16 v2bf;
typedef __attribute__((ext_vector_type(8)))  __bf16 v8bf;
typedef __attribute__((ext_vector_type(16))) __bf16 v16bf;
typedef __attribute__((ext_vector_type(8)))  float  v8f;
typedef __attribute__((ext_vector_type(4)))  unsigned int u32x4;
typedef __attribute__((ext_vector_type(4)))  int i32x4;
typedef __attribute__((ext_vector_type(8)))  int i32x8;

static constexpr int Bb = 8, T = 2048, Cc = 1024, HS = 1024;
static constexpr int M_TOT = Bb * T; // 16384

union FragB { v16bf v; v8bf h[2]; };

#define WMMA_BF16(a, b, c) \
  __builtin_amdgcn_wmma_f32_16x16x32_bf16(false, (a), false, (b), (short)0, (c), false, false)

// ---------------------------------------------------------------------------
// Kernel 1: out[bf16, M x HS] = x[f32, M x C] @ w[f32, C x HS]
// 128x128 tile per 256-thread block (8 waves, 2x4), BK = 32.
// ---------------------------------------------------------------------------
__global__ __launch_bounds__(256) void qkv_proj_kernel(
    const float* __restrict__ x, const float* __restrict__ w,
    bf16_t* __restrict__ out)
{
  __shared__ bf16_t As[128 * 40];   // [m][k], pitch 40 (80B rows, 16B aligned)
  __shared__ bf16_t Bs[128 * 40];   // transposed: [n][k]

  const int tid  = threadIdx.x;
  const int lane = tid & 31;
  const int wv   = tid >> 5;
  const int g    = lane >> 4;
  const int ln16 = lane & 15;
  const int wm   = (wv & 1) * 64;   // wave row block
  const int wn   = (wv >> 1) * 32;  // wave col block
  const int bm0  = blockIdx.y * 128;
  const int bn0  = blockIdx.x * 128;

  v8f acc[4][2];
  for (int i = 0; i < 4; ++i)
    for (int j = 0; j < 2; ++j)
      for (int e = 0; e < 8; ++e) acc[i][j][e] = 0.0f;

  for (int kc = 0; kc < Cc / 32; ++kc) {
    // stage A tile 128x32: each thread converts+stores a k-pair (b32 DS store)
#pragma unroll
    for (int p = 0; p < 8; ++p) {
      int m = p * 16 + (tid >> 4);
      int k = (tid & 15) * 2;
      float2 xv = *(const float2*)&x[(size_t)(bm0 + m) * Cc + kc * 32 + k];
      *(v2bf*)&As[m * 40 + k] = (v2bf){(bf16_t)xv.x, (bf16_t)xv.y};
    }
    // stage B tile 32x128 transposed: k-pair per thread (b32 DS store)
#pragma unroll
    for (int p = 0; p < 8; ++p) {
      int n = tid & 127;
      int k = (p * 2 + (tid >> 7)) * 2;
      float w0 = w[(size_t)(kc * 32 + k) * HS + bn0 + n];
      float w1 = w[(size_t)(kc * 32 + k + 1) * HS + bn0 + n];
      *(v2bf*)&Bs[n * 40 + k] = (v2bf){(bf16_t)w0, (bf16_t)w1};
    }
    __syncthreads();

    FragB a[4], bf[2];
#pragma unroll
    for (int tm = 0; tm < 4; ++tm) {
      const bf16_t* rp = &As[(wm + tm * 16 + ln16) * 40];
      a[tm].h[0] = *(const v8bf*)(rp + g * 8);
      a[tm].h[1] = *(const v8bf*)(rp + 16 + g * 8);
    }
#pragma unroll
    for (int tn = 0; tn < 2; ++tn) {
      const bf16_t* rp = &Bs[(wn + tn * 16 + ln16) * 40];
      bf[tn].h[0] = *(const v8bf*)(rp + g * 8);
      bf[tn].h[1] = *(const v8bf*)(rp + 16 + g * 8);
    }
#pragma unroll
    for (int tm = 0; tm < 4; ++tm)
#pragma unroll
      for (int tn = 0; tn < 2; ++tn)
        acc[tm][tn] = WMMA_BF16(a[tm].v, bf[tn].v, acc[tm][tn]);
    __syncthreads();
  }

  // epilogue: C/D layout M = r + 8*g, N = ln16
#pragma unroll
  for (int tm = 0; tm < 4; ++tm)
#pragma unroll
    for (int tn = 0; tn < 2; ++tn)
#pragma unroll
      for (int r = 0; r < 8; ++r) {
        int row = bm0 + wm + tm * 16 + r + g * 8;
        int col = bn0 + wn + tn * 16 + ln16;
        out[(size_t)row * HS + col] = (bf16_t)acc[tm][tn][r];
      }
}

// ---------------------------------------------------------------------------
// Kernel 2: fused causal attention for one (batch, 16-row query tile).
// Q tile staged by the Tensor Data Mover; 16 x T score block in LDS (f32),
// exact softmax, P (bf16) in LDS, O = P @ V with V staged transposed per
// 32-key block.
// ---------------------------------------------------------------------------
static constexpr int S_PITCH = 2064;  // floats, 16B-aligned rows

__global__ __launch_bounds__(256) void attn_kernel(
    const bf16_t* __restrict__ qb, const bf16_t* __restrict__ kb,
    const bf16_t* __restrict__ vb, float* __restrict__ out)
{
  extern __shared__ char smem[];
  float*  Ssh = (float*)smem;                                   // 16*2064*4 = 132096
  bf16_t* Psh = (bf16_t*)(smem + 16 * S_PITCH * 4);             // 16*2048*2 = 65536
  bf16_t* Wsh = (bf16_t*)(smem + 16 * S_PITCH * 4 + 65536);     // Q tile / Vt union: 65536
  float*  red  = (float*)(smem + 16 * S_PITCH * 4 + 2 * 65536); // 16*16
  float*  Mrow = red + 256;                                     // 16
  float*  Lrow = Mrow + 16;                                     // 16

  const int tid  = threadIdx.x;
  const int lane = tid & 31;
  const int wv   = tid >> 5;
  const int g    = lane >> 4;
  const int ln16 = lane & 15;
  const int qi   = blockIdx.x;
  const int b    = blockIdx.y;
  const int r0   = qi * 16;
  const int kv_len  = (qi + 1) * 16;
  const int k_steps = (kv_len + 31) / 32;
  const int kv_pad  = k_steps * 32;

  const bf16_t* Q = qb + (size_t)b * T * HS;
  const bf16_t* K = kb + (size_t)b * T * HS;
  const bf16_t* V = vb + (size_t)b * T * HS;

  // ---- stage Q tile (16 x 1024 bf16 = 32KB) into Wsh via TDM ----
  if (tid < 32) {   // one wave issues the DMA (TDM ignores EXEC; wave-level op)
    unsigned lds_off = (unsigned)(uintptr_t)Wsh;   // low 32 bits = LDS byte addr
    unsigned long long ga = (unsigned long long)(uintptr_t)(Q + (size_t)r0 * HS);
    // D# group 0: count=1, lds_addr, 57-bit global addr, type=2 ("image")
    u32x4 g0 = {1u, lds_off, (unsigned)ga,
                ((unsigned)(ga >> 32) & 0x01FFFFFFu) | 0x80000000u};
    // D# group 1: data_size=2B; tensor_dim0=1024, tensor_dim1=16;
    // tile_dim0=1024, tile_dim1=16; tensor_dim0_stride=1024
    i32x8 g1 = {(int)(1u << 16),
                (int)(1024u << 16),
                (int)(16u << 16),
                (int)(1024u << 16),
                16, 1024, 0, 0};
    i32x4 g2 = {0, 0, 0, 0};
    i32x4 g3 = {0, 0, 0, 0};
    i32x8 g4 = {0, 0, 0, 0, 0, 0, 0, 0};
    __builtin_amdgcn_tensor_load_to_lds(g0, g1, g2, g3, g4, 0);
    __builtin_amdgcn_s_wait_tensorcnt(0);
  }
  __syncthreads();

  // ---- phase 1: S = scale * Q K^T with causal mask, into LDS ----
  const float scale = 0.03125f;  // C^-0.5 = 1024^-0.5
  for (int ct = wv; ct <= qi; ct += 8) {
    v8f acc;
    for (int e = 0; e < 8; ++e) acc[e] = 0.0f;
    const bf16_t* qrow = Wsh + ln16 * 1024;                 // A: row m = ln16
    const bf16_t* krow = K + (size_t)(ct * 16 + ln16) * HS; // B: col n = ln16, contiguous k
    __builtin_prefetch(krow, 0, 3);
    for (int hc = 0; hc < HS / 32; ++hc) {
      FragB a, bf;
      a.h[0]  = *(const v8bf*)(qrow + hc * 32 + g * 8);
      a.h[1]  = *(const v8bf*)(qrow + hc * 32 + 16 + g * 8);
      bf.h[0] = *(const v8bf*)(krow + hc * 32 + g * 8);
      bf.h[1] = *(const v8bf*)(krow + hc * 32 + 16 + g * 8);
      acc = WMMA_BF16(a.v, bf.v, acc);
    }
#pragma unroll
    for (int r = 0; r < 8; ++r) {
      int m   = r + g * 8;
      int col = ct * 16 + ln16;
      float s = acc[r] * scale;
      if (col > r0 + m) s = -1.0e30f;
      Ssh[m * S_PITCH + col] = s;
    }
  }
  __syncthreads();

  // ---- phase 2: exact row softmax; P = exp(S - max) as bf16, keep l ----
  {
    int row = tid >> 4;
    int w16 = tid & 15;
    float mp = -1.0e30f;
    for (int c = w16; c < kv_len; c += 16) mp = fmaxf(mp, Ssh[row * S_PITCH + c]);
    red[row * 16 + w16] = mp;
    __syncthreads();
    if (w16 == 0) {
      float m = red[row * 16];
      for (int i = 1; i < 16; ++i) m = fmaxf(m, red[row * 16 + i]);
      Mrow[row] = m;
    }
    __syncthreads();
    float m = Mrow[row];
    float lp = 0.0f;
    for (int c = w16; c < kv_len; c += 16) {
      float e = __expf(Ssh[row * S_PITCH + c] - m);
      Psh[row * 2048 + c] = (bf16_t)e;
      lp += e;
    }
    if (kv_len + w16 < kv_pad) Psh[row * 2048 + kv_len + w16] = (bf16_t)0.0f;
    red[row * 16 + w16] = lp;
    __syncthreads();
    if (w16 == 0) {
      float l = 0.0f;
      for (int i = 0; i < 16; ++i) l += red[row * 16 + i];
      Lrow[row] = l;
    }
  }
  __syncthreads();

  // ---- phase 3: O = P @ V. Each wave owns 8 N-tiles: nb = (wv + 8t)*16 ----
  v8f oacc[8];
  for (int t = 0; t < 8; ++t)
    for (int e = 0; e < 8; ++e) oacc[t][e] = 0.0f;

  for (int ks = 0; ks < k_steps; ++ks) {
    __syncthreads();  // previous Vt (or Q) fully consumed
    // stage V[32 keys x 1024] transposed into Wsh as Vt[n*32 + k];
    // process key-row pairs so LDS writes are b32.
    for (int i = tid; i < 2048; i += 256) {
      int k2 = (i >> 7) * 2;
      int n8 = (i & 127) * 8;
      v8bf v0 = *(const v8bf*)(V + (size_t)(ks * 32 + k2) * HS + n8);
      v8bf v1 = *(const v8bf*)(V + (size_t)(ks * 32 + k2 + 1) * HS + n8);
#pragma unroll
      for (int e = 0; e < 8; ++e)
        *(v2bf*)&Wsh[(n8 + e) * 32 + k2] = (v2bf){v0[e], v1[e]};
    }
    __syncthreads();

    FragB a;  // A-fragment from P, loaded once, reused for all 8 N-tiles
    const bf16_t* prow = Psh + ln16 * 2048 + ks * 32;
    a.h[0] = *(const v8bf*)(prow + g * 8);
    a.h[1] = *(const v8bf*)(prow + 16 + g * 8);
#pragma unroll
    for (int t = 0; t < 8; ++t) {
      int nb = (wv + 8 * t) * 16;
      FragB bf;
      const bf16_t* vrow = Wsh + (size_t)(nb + ln16) * 32;
      bf.h[0] = *(const v8bf*)(vrow + g * 8);
      bf.h[1] = *(const v8bf*)(vrow + 16 + g * 8);
      oacc[t] = WMMA_BF16(a.v, bf.v, oacc[t]);
    }
  }

  // ---- epilogue: normalize by 1/l and store f32 ----
#pragma unroll
  for (int t = 0; t < 8; ++t) {
    int nb = (wv + 8 * t) * 16;
#pragma unroll
    for (int r = 0; r < 8; ++r) {
      int m = r + g * 8;
      out[((size_t)b * T + r0 + m) * HS + nb + ln16] = oacc[t][r] / Lrow[m];
    }
  }
}

// ---------------------------------------------------------------------------
extern "C" void kernel_launch(void* const* d_in, const int* in_sizes, int n_in,
                              void* d_out, int out_size, void* d_ws, size_t ws_size,
                              hipStream_t stream) {
  (void)in_sizes; (void)n_in; (void)out_size; (void)ws_size;
  const float* x     = (const float*)d_in[0];
  const float* k_wei = (const float*)d_in[1];
  const float* q_wei = (const float*)d_in[2];
  const float* v_wei = (const float*)d_in[3];
  // d_in[4] = mask (tril) -> implemented analytically as col<=row

  bf16_t* qb = (bf16_t*)d_ws;
  bf16_t* kb = qb + (size_t)M_TOT * HS;
  bf16_t* vb = kb + (size_t)M_TOT * HS;   // ws usage: 96 MB of bf16 q/k/v

  dim3 gp(HS / 128, M_TOT / 128);
  qkv_proj_kernel<<<gp, 256, 0, stream>>>(x, q_wei, qb);
  qkv_proj_kernel<<<gp, 256, 0, stream>>>(x, k_wei, kb);
  qkv_proj_kernel<<<gp, 256, 0, stream>>>(x, v_wei, vb);

  size_t smem = (size_t)16 * S_PITCH * 4 + 65536 + 65536 + 256 * 4 + 32 * 4; // ~258 KB (CDNA5 WGP LDS)
  (void)hipFuncSetAttribute((const void*)attn_kernel,
                            hipFuncAttributeMaxDynamicSharedMemorySize, (int)smem);
  attn_kernel<<<dim3(T / 16, Bb), 256, smem, stream>>>(qb, kb, vb, (float*)d_out);
}
